// GNNModule_12661563588934
// MI455X (gfx1250) — matrix operand order repaired and checked by reading, not compiled, not run on hardware
//
#include <hip/hip_runtime.h>

// PNAConv (5 layers, divide_input, aggr = mean/sum/max) for MI455X / gfx1250.
// wave32 + v_wmma_f32_16x16x32_f16 for all matrix work. Linear-layer folding
// (done ONCE per layer by a tiny fold kernel, consumed by all blocks):
//  - edge encoder folded into per-tower pre-MLP weights (K = 8+8+16 = 32 exact)
//  - mixing linear folded into per-tower post-MLP weights (chained WMMA accum)
// Branchless u32-keyed float max atomics + native f32 add atomics for scatter;
// out-of-range edges scatter to a dummy row (index N) -> no per-element guards.

typedef __attribute__((ext_vector_type(16))) _Float16 v16h;
typedef __attribute__((ext_vector_type(8)))  _Float16 v8h;
typedef __attribute__((ext_vector_type(8)))  float    v8f;

#define T_  4
#define FT_ 8
#define F_  32
#define ED_ 16
#define L_  5
#define WAVES 4   // 128-thread blocks = 4 wave32

// ---- monotone f32 <-> u32 key (float order == unsigned order) --------------
__device__ __forceinline__ unsigned f32_key(float v) {
  unsigned u = __float_as_uint(v);
  unsigned mask = (unsigned)(((int)u) >> 31) | 0x80000000u;
  return u ^ mask;  // v_ashrrev + v_bitop3
}
__device__ __forceinline__ float key_f32(unsigned k) {
  unsigned mask = 0x80000000u | ~(unsigned)(((int)k) >> 31);
  return __uint_as_float(k ^ mask);  // branchless inverse
}

// ---- WMMA fragment helpers -------------------------------------------------
// A-matrix 16x32 f16 layout (ISA 7.12.2): lanes 0-15 row M=lane, halves
// [K0..7 | K16..23]; lanes 16-31 row M=lane-16, halves [K8..15 | K24..31].
__device__ __forceinline__ v16h load_a_frag(const _Float16* tile /*[16][32]*/,
                                            int m, int hi) {
  const _Float16* row = tile + m * 32;
  v8h lo = *(const v8h*)(row + hi * 8);
  v8h hh = *(const v8h*)(row + 16 + hi * 8);
  return __builtin_shufflevector(lo, hh, 0, 1, 2, 3, 4, 5, 6, 7,
                                 8, 9, 10, 11, 12, 13, 14, 15);
}

// B-matrix 32x16 f16: lane n<16 holds column n, K=[0..7 | 16..23]; lanes 16-31
// hold column n-16, K=[8..15 | 24..31] (two K-values per VGPR).
__device__ __forceinline__ v16h load_b_frag(const _Float16* B, int rowStride,
                                            int colOff, int lane) {
  const int nn = (lane & 15) + colOff;
  const int kb = (lane >> 4) * 8;
  v16h b;
#pragma unroll
  for (int i = 0; i < 16; ++i) b[i] = B[(kb + i + (i >= 8 ? 8 : 0)) * rowStride + nn];
  return b;
}

__device__ __forceinline__ v8h cvt8(const float* f) {
  v8h o;
#pragma unroll
  for (int i = 0; i < 8; ++i) o[i] = (_Float16)f[i];
  return o;
}

// ---- small utility kernels -------------------------------------------------
__global__ void init_cnt_kernel(int* cnt, int n) {
  int i = blockIdx.x * blockDim.x + threadIdx.x;
  if (i < n) cnt[i] = 0;
}

__global__ void degree_kernel(const int* __restrict__ dst, int* __restrict__ cnt, int E) {
  for (int e = blockIdx.x * blockDim.x + threadIdx.x; e < E;
       e += gridDim.x * blockDim.x)
    atomicAdd(&cnt[dst[e]], 1);
}

__global__ void init_aggr_kernel(float* __restrict__ ssum,
                                 unsigned* __restrict__ smaxKey, int total) {
  int i = blockIdx.x * blockDim.x + threadIdx.x;
  if (i < total) { ssum[i] = 0.0f; smaxKey[i] = 0u; }  // key 0 == -inf domain
}

// ---- one-shot weight folding (per layer) -----------------------------------
// bpreF[t][32][16]: rows 0..7 W_xi, 8..15 W_xj, 16..31 eW @ W_ee (fp32->f16)
// pbF[t*8+f]      : pb + eb @ W_ee
// qF[t][32][32]   : poW_t @ lW[t*8 : t*8+8]
// cbF[nn]         : lb + pob @ lW
__global__ void fold_weights_kernel(
    const float* __restrict__ eW, const float* __restrict__ eb,
    const float* __restrict__ pW, const float* __restrict__ pb,
    const float* __restrict__ poW, const float* __restrict__ pob,
    const float* __restrict__ linW, const float* __restrict__ linb,
    _Float16* __restrict__ bpreF, float* __restrict__ pbF,
    _Float16* __restrict__ qF, float* __restrict__ cbF) {
  const int gid = blockIdx.x * blockDim.x + threadIdx.x;
  const int gsz = gridDim.x * blockDim.x;

  for (int idx = gid; idx < T_ * 32 * 16; idx += gsz) {
    int t = idx >> 9, rem = idx & 511, k = rem >> 4, nn = rem & 15;
    float v = 0.0f;
    if (nn < FT_) {
      if (k < 2 * FT_) {
        v = pW[t * (3 * FT_ * FT_) + k * FT_ + nn];
      } else {
        const int ka = k - 2 * FT_;  // edge_attr dim 0..15
        float acc = 0.0f;
#pragma unroll
        for (int j = 0; j < FT_; ++j)
          acc = fmaf(eW[ka * FT_ + j],
                     pW[t * (3 * FT_ * FT_) + (2 * FT_ + j) * FT_ + nn], acc);
        v = acc;
      }
    }
    bpreF[idx] = (_Float16)v;
  }
  for (int idx = gid; idx < T_ * FT_; idx += gsz) {
    int t = idx >> 3, f = idx & 7;
    float acc = pb[idx];
#pragma unroll
    for (int j = 0; j < FT_; ++j)
      acc = fmaf(eb[j], pW[t * (3 * FT_ * FT_) + (2 * FT_ + j) * FT_ + f], acc);
    pbF[idx] = acc;
  }
  for (int idx = gid; idx < T_ * 32 * F_; idx += gsz) {
    int t = idx >> 10, rem = idx & 1023, k = rem >> 5, nn = rem & 31;
    float acc = 0.0f;
#pragma unroll
    for (int f = 0; f < FT_; ++f)
      acc = fmaf(poW[t * (4 * FT_ * FT_) + k * FT_ + f],
                 linW[(t * FT_ + f) * F_ + nn], acc);
    qF[idx] = (_Float16)acc;
  }
  for (int idx = gid; idx < F_; idx += gsz) {
    float acc = linb[idx];
    for (int j = 0; j < T_ * FT_; ++j)
      acc = fmaf(pob[j], linW[j * F_ + idx], acc);
    cbF[idx] = acc;
  }
}

// ---- edge kernel: gather + folded pre-MLP (WMMA) + atomic scatter ----------
struct EdgeSmem {
  _Float16 Bpre[T_][32][16];       // folded weights [Wxi;Wxj;eW@Wee] (N8->16)
  _Float16 As[WAVES][T_][16][32];  // per-wave tower A tiles: 16 edges x 32 K
  float    pb[T_ * FT_];           // folded bias pb + eb@Wee
  int      dsts[WAVES][16];        // pre-scaled dst*F_ (dummy N*F_ if invalid)
};

__global__ __launch_bounds__(128) void pna_edge_kernel(
    const float* __restrict__ x, const int* __restrict__ srcIdx,
    const int* __restrict__ dstIdx, const float* __restrict__ edge_attr,
    const _Float16* __restrict__ bpreF, const float* __restrict__ pbF,
    float* __restrict__ ssum, unsigned* __restrict__ smaxKey, int E, int N) {
  __shared__ EdgeSmem sh;
  const int tid = threadIdx.x;

  // Stage pre-folded weights (vector copy: 2 iters/thread).
  for (int idx = tid; idx < T_ * 32 * 16 / 8; idx += blockDim.x)
    ((v8h*)&sh.Bpre[0][0][0])[idx] = ((const v8h*)bpreF)[idx];
  if (tid < T_ * FT_) sh.pb[tid] = pbF[tid];
  __syncthreads();

  const int lane = tid & 31;
  const int wave = tid >> 5;
  const int m  = lane & 15;   // edge slot / A row
  const int hi = lane >> 4;   // lane half
  const int n  = lane & 15;   // C column

  v16h bpre[T_];
#pragma unroll
  for (int t = 0; t < T_; ++t) bpre[t] = load_b_frag(&sh.Bpre[t][0][0], 16, 0, lane);
  float pbc[T_];
#pragma unroll
  for (int t = 0; t < T_; ++t) pbc[t] = sh.pb[t * FT_ + (n & 7)];

  const int numTiles   = (E + 15) >> 4;
  const int waveId     = blockIdx.x * WAVES + wave;
  const int waveStride = gridDim.x * WAVES;

  for (int tile = waveId; tile < numTiles; tile += waveStride) {
    const int base  = tile << 4;
    const int e     = base + m;
    const bool valid = (e < E);
    int s = 0, d = 0;
    if (valid) { s = srcIdx[e]; d = dstIdx[e]; }
    // Invalid edges scatter to dummy row N -> branchless atomics below.
    if (hi == 0) sh.dsts[wave][m] = (valid ? d : N) * F_;

    // Prefetch the next tile's edge_attr rows (global_prefetch_b8).
    {
      int enext = base + waveStride * 16 + m;
      if (enext < E) __builtin_prefetch(edge_attr + (size_t)enext * ED_, 0, 1);
    }

    // Per-edge attr row as f16 (both lane halves need it for their towers).
    float ea[ED_];
    if (valid) {
      const float4* pea = (const float4*)(edge_attr + (size_t)e * ED_);
#pragma unroll
      for (int q = 0; q < 4; ++q) {
        float4 f = pea[q];
        ea[q * 4 + 0] = f.x; ea[q * 4 + 1] = f.y;
        ea[q * 4 + 2] = f.z; ea[q * 4 + 3] = f.w;
      }
    } else {
#pragma unroll
      for (int q = 0; q < ED_; ++q) ea[q] = 0.0f;
    }
    const v8h ah0 = cvt8(ea);
    const v8h ah1 = cvt8(ea + 8);

    // Stage tower A rows: [xi_t(8) | xj_t(8) | attr(16)] -- K=32, no padding.
    // Lane-half hi handles towers 2*hi and 2*hi+1; 128-bit LDS stores.
    const int t0 = hi * 2;
#pragma unroll
    for (int tt = 0; tt < 2; ++tt) {
      const int t = t0 + tt;
      _Float16* row = &sh.As[wave][t][m][0];
      float fd[8], fs[8];
      if (valid) {
        const float4* pd = (const float4*)(x + (size_t)d * F_ + t * FT_);
        const float4* ps = (const float4*)(x + (size_t)s * F_ + t * FT_);
        float4 d0 = pd[0], d1 = pd[1], s0 = ps[0], s1 = ps[1];
        fd[0]=d0.x; fd[1]=d0.y; fd[2]=d0.z; fd[3]=d0.w;
        fd[4]=d1.x; fd[5]=d1.y; fd[6]=d1.z; fd[7]=d1.w;
        fs[0]=s0.x; fs[1]=s0.y; fs[2]=s0.z; fs[3]=s0.w;
        fs[4]=s1.x; fs[5]=s1.y; fs[6]=s1.z; fs[7]=s1.w;
      } else {
#pragma unroll
        for (int i = 0; i < 8; ++i) { fd[i] = 0.0f; fs[i] = 0.0f; }
      }
      *(v8h*)(row + 0)  = cvt8(fd);
      *(v8h*)(row + 8)  = cvt8(fs);
      *(v8h*)(row + 16) = ah0;
      *(v8h*)(row + 24) = ah1;
    }
    __builtin_amdgcn_wave_barrier();  // LDS in-order per wave; fence compiler

    // Preload this lane's 8 pre-scaled destination offsets (reused per tower).
    int obase[8];
    {
      const int4* dp = (const int4*)&sh.dsts[wave][hi * 8];
      int4 q0 = dp[0], q1 = dp[1];
      obase[0]=q0.x; obase[1]=q0.y; obase[2]=q0.z; obase[3]=q0.w;
      obase[4]=q1.x; obase[5]=q1.y; obase[6]=q1.z; obase[7]=q1.w;
    }

    // One WMMA per tower: h[16 edges x 8ch] = A(16x32) @ Bpre'(32x16) + bias'.
#pragma unroll
    for (int t = 0; t < T_; ++t) {
      v16h a = load_a_frag(&sh.As[wave][t][0][0], m, hi);
      v8f c;
#pragma unroll
      for (int r = 0; r < 8; ++r) c[r] = pbc[t];  // bias pre-broadcast into C
      c = __builtin_amdgcn_wmma_f32_16x16x32_f16(false, a, false, bpre[t],
                                                 (short)0, c, false, false);
      if (n < FT_) {
#pragma unroll
        for (int r = 0; r < 8; ++r) {
          const float v = c[r];
          const int off = obase[r] + t * FT_ + n;
          atomicAdd(ssum + off, v);                 // global_atomic_add_f32
          atomicMax(smaxKey + off, f32_key(v));     // global_atomic_max_u32
        }
      }
    }
    __builtin_amdgcn_wave_barrier();  // before next iteration reuses tiles
  }
}

// ---- node kernel: folded post-MLP x mixing linear (chained WMMA) + ReLU ----
struct NodeSmem {
  _Float16 Q[T_][32][32];          // folded Q_t = poW_t @ lW[t*8 : t*8+8]
  _Float16 Ain[WAVES][T_][16][32]; // [xt|mean|sum|max] per tower, K=32 exact
  float    cb[F_];                 // folded bias lb + pob @ lW
};

__global__ __launch_bounds__(128) void pna_node_kernel(
    const float* __restrict__ x, const float* __restrict__ ssum,
    const unsigned* __restrict__ smaxKey, const int* __restrict__ cnt,
    const _Float16* __restrict__ qF, const float* __restrict__ cbF,
    float* __restrict__ out, int N) {
  __shared__ NodeSmem sh;
  const int tid = threadIdx.x;

  // Stage pre-folded weights (vector copy: 4 iters/thread).
  for (int idx = tid; idx < T_ * 32 * F_ / 8; idx += blockDim.x)
    ((v8h*)&sh.Q[0][0][0])[idx] = ((const v8h*)qF)[idx];
  if (tid < F_) sh.cb[tid] = cbF[tid];
  __syncthreads();

  const int lane = tid & 31;
  const int wave = tid >> 5;
  const int m  = lane & 15;
  const int hi = lane >> 4;
  const int n  = lane & 15;

  v16h q0[T_], q1[T_];
#pragma unroll
  for (int t = 0; t < T_; ++t) {
    q0[t] = load_b_frag(&sh.Q[t][0][0], F_, 0, lane);
    q1[t] = load_b_frag(&sh.Q[t][0][0], F_, 16, lane);
  }
  const float cb0 = sh.cb[n];
  const float cb1 = sh.cb[16 + n];

  const int numTiles   = (N + 15) >> 4;
  const int waveId     = blockIdx.x * WAVES + wave;
  const int waveStride = gridDim.x * WAVES;

  for (int tile = waveId; tile < numTiles; tile += waveStride) {
    const int base  = tile << 4;
    const int node  = base + m;
    const bool valid = (node < N);
    float cv  = valid ? (float)cnt[node] : 0.0f;
    float inv = 1.0f / fmaxf(cv, 1.0f);
    bool  has = cv > 0.0f;

    const int t0 = hi * 2;
#pragma unroll
    for (int tt = 0; tt < 2; ++tt) {
      const int t = t0 + tt;
      _Float16* row = &sh.Ain[wave][t][m][0];
      float xv[8], sv[8], mv[8], av[8];
      if (valid) {
        const size_t o = (size_t)node * F_ + t * FT_;
        const float4* px = (const float4*)(x + o);
        const float4* ps = (const float4*)(ssum + o);
        const uint4*  pm = (const uint4*)(smaxKey + o);
        float4 a0 = px[0], a1 = px[1], b0 = ps[0], b1 = ps[1];
        uint4  c0 = pm[0], c1 = pm[1];
        xv[0]=a0.x; xv[1]=a0.y; xv[2]=a0.z; xv[3]=a0.w;
        xv[4]=a1.x; xv[5]=a1.y; xv[6]=a1.z; xv[7]=a1.w;
        sv[0]=b0.x; sv[1]=b0.y; sv[2]=b0.z; sv[3]=b0.w;
        sv[4]=b1.x; sv[5]=b1.y; sv[6]=b1.z; sv[7]=b1.w;
        mv[0]=key_f32(c0.x); mv[1]=key_f32(c0.y);
        mv[2]=key_f32(c0.z); mv[3]=key_f32(c0.w);
        mv[4]=key_f32(c1.x); mv[5]=key_f32(c1.y);
        mv[6]=key_f32(c1.z); mv[7]=key_f32(c1.w);
      } else {
#pragma unroll
        for (int i = 0; i < 8; ++i) { xv[i]=0.f; sv[i]=0.f; mv[i]=0.f; }
      }
#pragma unroll
      for (int i = 0; i < 8; ++i) av[i] = sv[i] * inv;            // mean
      float mx[8];
#pragma unroll
      for (int i = 0; i < 8; ++i) mx[i] = has ? mv[i] : 0.0f;     // max (empty->0)
      *(v8h*)(row + 0)  = cvt8(xv);
      *(v8h*)(row + 8)  = cvt8(av);
      *(v8h*)(row + 16) = cvt8(sv);
      *(v8h*)(row + 24) = cvt8(mx);
    }
    __builtin_amdgcn_wave_barrier();

    // z[16x32] = sum_t Ain_t(16x32) @ Q_t(32x32) + cb  (chained WMMA accum)
    v8f c0, c1;
#pragma unroll
    for (int r = 0; r < 8; ++r) { c0[r] = cb0; c1[r] = cb1; }
#pragma unroll
    for (int t = 0; t < T_; ++t) {
      v16h a = load_a_frag(&sh.Ain[wave][t][0][0], m, hi);
      c0 = __builtin_amdgcn_wmma_f32_16x16x32_f16(false, a, false, q0[t],
                                                  (short)0, c0, false, false);
      c1 = __builtin_amdgcn_wmma_f32_16x16x32_f16(false, a, false, q1[t],
                                                  (short)0, c1, false, false);
    }
    // ReLU + store (columns n and 16+n; rows 8*hi+r).
#pragma unroll
    for (int r = 0; r < 8; ++r) {
      const int nd = base + hi * 8 + r;
      if (nd < N) {
        out[(size_t)nd * F_ + n]      = fmaxf(c0[r], 0.0f);
        out[(size_t)nd * F_ + 16 + n] = fmaxf(c1[r], 0.0f);
      }
    }
    __builtin_amdgcn_wave_barrier();
  }
}

// ---- host-side launcher ----------------------------------------------------
extern "C" void kernel_launch(void* const* d_in, const int* in_sizes, int n_in,
                              void* d_out, int out_size, void* d_ws, size_t ws_size,
                              hipStream_t stream) {
  (void)n_in; (void)out_size; (void)ws_size;
  const float* x_in   = (const float*)d_in[0];
  const int*   eidx   = (const int*)d_in[1];
  const float* eattr  = (const float*)d_in[2];
  const float* edge_W = (const float*)d_in[3];
  const float* edge_b = (const float*)d_in[4];
  const float* pre_W  = (const float*)d_in[5];
  const float* pre_b  = (const float*)d_in[6];
  const float* post_W = (const float*)d_in[7];
  const float* post_b = (const float*)d_in[8];
  const float* lin_W  = (const float*)d_in[9];
  const float* lin_b  = (const float*)d_in[10];

  const int N = in_sizes[0] / F_;
  const int E = in_sizes[1] / 2;
  const int* srcIdx = eidx;
  const int* dstIdx = eidx + E;

  // ssum/smax carry one extra dummy row (index N) absorbing padded edges.
  float*    xb0  = (float*)d_ws;
  float*    xb1  = xb0  + (size_t)N * F_;
  float*    ssum = xb1  + (size_t)N * F_;
  unsigned* smax = (unsigned*)(ssum + (size_t)(N + 1) * F_);
  int*      cnt  = (int*)(smax + (size_t)(N + 1) * F_);
  // Per-layer folded weight buffers (reused across layers; 16B aligned).
  uintptr_t p = (uintptr_t)(cnt + N);
  p = (p + 15u) & ~(uintptr_t)15u;
  _Float16* bpreF = (_Float16*)p;  p += (size_t)T_ * 32 * 16 * sizeof(_Float16);
  _Float16* qF    = (_Float16*)p;  p += (size_t)T_ * 32 * F_ * sizeof(_Float16);
  float*    pbF   = (float*)p;     p += (size_t)T_ * FT_ * sizeof(float);
  float*    cbF   = (float*)p;

  // Degree counts (dst-only, identical across layers): compute once per launch.
  init_cnt_kernel<<<(N + 255) / 256, 256, 0, stream>>>(cnt, N);
  degree_kernel<<<2048, 256, 0, stream>>>(dstIdx, cnt, E);

  const float* xcur = x_in;
  for (int l = 0; l < L_; ++l) {
    const int total = (N + 1) * F_;
    init_aggr_kernel<<<(total + 255) / 256, 256, 0, stream>>>(ssum, smax, total);

    // One-shot fold of this layer's weights (enough threads for 1 trip/loop).
    fold_weights_kernel<<<16, 256, 0, stream>>>(
        edge_W + (size_t)l * ED_ * FT_,
        edge_b + (size_t)l * FT_,
        pre_W  + (size_t)l * T_ * 3 * FT_ * FT_,
        pre_b  + (size_t)l * T_ * FT_,
        post_W + (size_t)l * T_ * 4 * FT_ * FT_,
        post_b + (size_t)l * T_ * FT_,
        lin_W  + (size_t)l * F_ * F_,
        lin_b  + (size_t)l * F_,
        bpreF, pbF, qF, cbF);

    pna_edge_kernel<<<2048, 128, 0, stream>>>(
        xcur, srcIdx, dstIdx, eattr, bpreF, pbF, ssum, smax, E, N);

    float* xout = (l == L_ - 1) ? (float*)d_out : ((l & 1) ? xb1 : xb0);
    pna_node_kernel<<<512, 128, 0, stream>>>(
        xcur, ssum, smax, cnt, qF, cbF, xout, N);
    xcur = xout;
  }
}
